// VisionPatchAttention_57672820850759
// MI455X (gfx1250) — matrix operand order repaired and checked by reading, not compiled, not run on hardware
//
#include <hip/hip_runtime.h>
#include <hip/hip_bf16.h>

// ---------------------------------------------------------------------------
// VisionPatchAttention for MI455X (gfx1250, wave32, WMMA 16x16x32 f16)
//   B=32, N=576, T=577 tokens, C=768, H=12 heads, HD=64, MP=24
//   Keys padded 577 -> 608 (38 tiles of 16 / 19 WMMA-K chunks of 32)
// ---------------------------------------------------------------------------

#define BB 32
#define NN 576
#define TT 577
#define CC 768
#define HH 12
#define HD 64
#define MM (BB * TT)      // 18464 rows (divisible by 32)
#define NK 608            // padded key/row count
#define QK_SCALE 0.125f   // 64^-0.5

typedef __attribute__((ext_vector_type(16))) _Float16 v16h;
typedef __attribute__((ext_vector_type(8)))  float    v8f;

union Frag { v16h h; uint4 q[2]; };

__device__ __forceinline__ v8f wmma_f16(const v16h& a, const v16h& b, v8f c) {
  return __builtin_amdgcn_wmma_f32_16x16x32_f16(false, a, false, b,
                                                (short)0, c, false, false);
}

// ---------------------------------------------------------------------------
// 1) prep: xu = [cls+pos ; depthwise3x3(x+pos)+conv_b]  -> f16 [MM][CC]
// ---------------------------------------------------------------------------
__global__ void prep_xu_kernel(const float* __restrict__ x,
                               const float* __restrict__ conv_w,
                               const float* __restrict__ conv_b,
                               const float* __restrict__ pos,
                               const float* __restrict__ cls,
                               _Float16* __restrict__ xu) {
  int idx = blockIdx.x * blockDim.x + threadIdx.x;
  if (idx >= MM * CC) return;
  int c = idx % CC;
  int m = idx / CC;
  int b = m / TT, t = m % TT;
  float val;
  if (t == 0) {
    val = cls[c] + pos[c];
  } else {
    int p = t - 1, i = p / 24, j = p % 24;
    float acc = conv_b[c];
#pragma unroll
    for (int di = -1; di <= 1; ++di)
#pragma unroll
      for (int dj = -1; dj <= 1; ++dj) {
        int ii = i + di, jj = j + dj;
        if (ii < 0 || ii >= 24 || jj < 0 || jj >= 24) continue;
        int pp = ii * 24 + jj;
        float xv = x[(size_t)(b * NN + pp) * CC + c] + pos[(size_t)(1 + pp) * CC + c];
        acc += xv * conv_w[c * 9 + (di + 1) * 3 + (dj + 1)];
      }
    val = acc;
  }
  xu[idx] = (_Float16)val;
}

// ---------------------------------------------------------------------------
// 2) transpose + fp16-convert weights: Wt[mat][n][k] = W[mat][k][n]
// ---------------------------------------------------------------------------
__global__ void prep_w_kernel(const float* __restrict__ Wq, const float* __restrict__ Wk,
                              const float* __restrict__ Wv, const float* __restrict__ Wo,
                              _Float16* __restrict__ Wt) {
  int idx = blockIdx.x * blockDim.x + threadIdx.x;
  if (idx >= 4 * CC * CC) return;
  int mat = idx / (CC * CC);
  int r = idx % (CC * CC);
  int n = r / CC, k = r % CC;
  const float* W = (mat == 0) ? Wq : (mat == 1) ? Wk : (mat == 2) ? Wv : Wo;
  Wt[idx] = (_Float16)W[(size_t)k * CC + n];
}

// ---------------------------------------------------------------------------
// 3) zero q/k/v pad region (whole buffers, uint4 stores)
// ---------------------------------------------------------------------------
__global__ void zero_kernel(uint4* __restrict__ p, long n4) {
  long idx = (long)blockIdx.x * blockDim.x + threadIdx.x;
  if (idx < n4) p[idx] = make_uint4(0u, 0u, 0u, 0u);
}

// ---------------------------------------------------------------------------
// 4) WMMA GEMM:  Y[MM][CC] = A[MM][CC](f16) @ Wt^T + bias, Wt is [n][k] f16.
//    2 waves / block, 32x64 tile; B fragments are two 16B global loads at k+8h and k+16+8h.
//    mode 0: f32 row-major to out. mode 1: f16 head layout [b*H+h][NK][64]*scale
// ---------------------------------------------------------------------------
__global__ void __launch_bounds__(64)
gemm_kernel(const _Float16* __restrict__ A, const _Float16* __restrict__ Wt,
            const float* __restrict__ bias, void* __restrict__ out,
            int mode, float scale) {
  int wave = threadIdx.x >> 5;
  int lane = threadIdx.x & 31;
  int l15 = lane & 15, hi = lane >> 4;
  int m0 = blockIdx.x * 32 + wave * 16;
  int n0 = blockIdx.y * 64;

  v8f acc[4] = {};
  const _Float16* arow = A + (size_t)(m0 + l15) * CC;

  for (int kc = 0; kc < CC / 32; ++kc) {
    Frag a;
    const _Float16* ap = arow + kc * 32 + 8 * hi;   // A layout: chunks K+0..7, K+16..23
    a.q[0] = *(const uint4*)(ap);
    a.q[1] = *(const uint4*)(ap + 16);
#pragma unroll
    for (int nb = 0; nb < 4; ++nb) {
      Frag b;
      const _Float16* bp = Wt + (size_t)(n0 + nb * 16 + l15) * CC + kc * 32 + 8 * hi;
      b.q[0] = *(const uint4*)(bp);
      b.q[1] = *(const uint4*)(bp + 16);
      acc[nb] = wmma_f16(a.h, b.h, acc[nb]);
    }
  }

#pragma unroll
  for (int nb = 0; nb < 4; ++nb) {
    int col = n0 + nb * 16 + l15;
    float bv = bias[col];
#pragma unroll
    for (int r = 0; r < 8; ++r) {
      int m = m0 + r + 8 * hi;
      float y = (acc[nb][r] + bv) * scale;
      if (mode == 0) {
        ((float*)out)[(size_t)m * CC + col] = y;
      } else {
        int b = m / TT, t = m % TT;
        int h = col >> 6, hd = col & 63;
        ((_Float16*)out)[(((size_t)(b * HH + h) * NK + t) << 6) + hd] = (_Float16)y;
      }
    }
  }
}

// ---------------------------------------------------------------------------
// 5) attention: per (qgroup, h, b). 4 waves, each owns a 16-row query tile.
//    LDS: V^T [64][608] f16 (shared) + per-wave scores [16][608] f32.
//    QK^T WMMA -> bias gather + softmax in LDS -> writes aw -> P@V WMMA.
// ---------------------------------------------------------------------------
__global__ void __launch_bounds__(128)
attn_kernel(const _Float16* __restrict__ q, const _Float16* __restrict__ k,
            const _Float16* __restrict__ v, const float* __restrict__ rel_table,
            const int* __restrict__ rel_index, float* __restrict__ aw,
            _Float16* __restrict__ xattn) {
  extern __shared__ __align__(16) char smem[];
  _Float16* vt = (_Float16*)smem;                       // [64][NK]
  float* allsc = (float*)(smem + (size_t)64 * NK * 2);  // 4 x [16][NK]

  int wave = threadIdx.x >> 5, lane = threadIdx.x & 31;
  int l15 = lane & 15, hi = lane >> 4;
  int h = blockIdx.y, b = blockIdx.z;
  int bh = b * HH + h;
  const _Float16* qb = q + (size_t)bh * NK * HD;
  const _Float16* kb = k + (size_t)bh * NK * HD;
  const _Float16* vb = v + (size_t)bh * NK * HD;

  // stage V^T into LDS (coalesced global reads)
  for (int idx = threadIdx.x; idx < NK * HD; idx += 128) {
    int key = idx >> 6, hd = idx & 63;
    vt[hd * NK + key] = vb[idx];
  }
  __syncthreads();

  int qt = blockIdx.x * 4 + wave;
  if (qt >= 37) return;
  int q0 = qt * 16;
  float* scores = allsc + (size_t)wave * 16 * NK;

  // Q fragments (two K-chunks of the 64-dim), direct global b128 loads
  Frag qa[2];
  const _Float16* qrow = qb + (size_t)(q0 + l15) * HD;
#pragma unroll
  for (int c = 0; c < 2; ++c) {
    const _Float16* qp = qrow + c * 32 + 8 * hi;
    qa[c].q[0] = *(const uint4*)(qp);
    qa[c].q[1] = *(const uint4*)(qp + 16);
  }

  // S = Q K^T over 38 key tiles
  for (int kt = 0; kt < NK / 16; ++kt) {
    v8f acc = {};
#pragma unroll
    for (int c = 0; c < 2; ++c) {
      Frag kf;
      const _Float16* kp = kb + (size_t)(kt * 16 + l15) * HD + c * 32 + 8 * hi;
      kf.q[0] = *(const uint4*)(kp);
      kf.q[1] = *(const uint4*)(kp + 16);
      acc = wmma_f16(qa[c].h, kf.h, acc);
    }
#pragma unroll
    for (int r = 0; r < 8; ++r)
      scores[(r + 8 * hi) * NK + kt * 16 + l15] = acc[r];
  }
  asm volatile("" ::: "memory");  // keep LDS store->load order at source level

  // softmax per row, add gathered rel-pos bias, emit aw (f32 output #2)
  size_t awbase = (size_t)bh * TT * TT;
  for (int r = 0; r < 16; ++r) {
    int qr = q0 + r;
    float mx = -1e30f;
    for (int col = lane; col < TT; col += 32) {
      float s = scores[r * NK + col];
      if (col > 0 && qr > 0 && qr < TT) {
        int ridx = rel_index[(size_t)(qr - 1) * NN + (col - 1)];
        s += rel_table[ridx * HH + h];
      }
      scores[r * NK + col] = s;
      mx = fmaxf(mx, s);
    }
#pragma unroll
    for (int o = 16; o > 0; o >>= 1) mx = fmaxf(mx, __shfl_xor(mx, o, 32));
    float sum = 0.f;
    for (int col = lane; col < TT; col += 32) {
      float e = __expf(scores[r * NK + col] - mx);
      scores[r * NK + col] = e;
      sum += e;
    }
#pragma unroll
    for (int o = 16; o > 0; o >>= 1) sum += __shfl_xor(sum, o, 32);
    float inv = 1.0f / sum;
    if (qr < TT) {
      for (int col = lane; col < TT; col += 32) {
        float p = scores[r * NK + col] * inv;
        scores[r * NK + col] = p;
        aw[awbase + (size_t)qr * TT + col] = p;
      }
    } else {
      for (int col = lane; col < TT; col += 32) scores[r * NK + col] *= inv;
    }
    for (int col = TT + lane; col < NK; col += 32) scores[r * NK + col] = 0.f;
  }
  asm volatile("" ::: "memory");

  // O = P @ V   (A from f32 scores in LDS with cvt, B from V^T in LDS)
  v8f o[4] = {};
  for (int kc = 0; kc < NK / 32; ++kc) {
    Frag pa;
    const float* p0 = scores + l15 * NK + kc * 32 + 8 * hi;
    const float* p1 = p0 + 16;
#pragma unroll
    for (int j = 0; j < 8; ++j) {
      pa.h[j]     = (_Float16)p0[j];
      pa.h[8 + j] = (_Float16)p1[j];
    }
#pragma unroll
    for (int nb = 0; nb < 4; ++nb) {
      Frag bf;
      const _Float16* vp = vt + (size_t)(nb * 16 + l15) * NK + kc * 32 + 8 * hi;
      bf.q[0] = *(const uint4*)(vp);
      bf.q[1] = *(const uint4*)(vp + 16);
      o[nb] = wmma_f16(pa.h, bf.h, o[nb]);
    }
  }

#pragma unroll
  for (int nb = 0; nb < 4; ++nb)
#pragma unroll
    for (int r = 0; r < 8; ++r) {
      int qr = q0 + r + 8 * hi;
      if (qr < TT) {
        int hd = nb * 16 + l15;
        xattn[(size_t)(b * TT + qr) * CC + h * HD + hd] = (_Float16)o[nb][r];
      }
    }
}

// ---------------------------------------------------------------------------
// launch
// ---------------------------------------------------------------------------
extern "C" void kernel_launch(void* const* d_in, const int* in_sizes, int n_in,
                              void* d_out, int out_size, void* d_ws, size_t ws_size,
                              hipStream_t stream) {
  const float* x      = (const float*)d_in[0];
  const float* Wq     = (const float*)d_in[1];
  const float* bq     = (const float*)d_in[2];
  const float* Wk     = (const float*)d_in[3];
  const float* bk     = (const float*)d_in[4];
  const float* Wv     = (const float*)d_in[5];
  const float* bv     = (const float*)d_in[6];
  const float* Wo     = (const float*)d_in[7];
  const float* bo     = (const float*)d_in[8];
  const float* conv_w = (const float*)d_in[9];
  const float* conv_b = (const float*)d_in[10];
  const float* pos    = (const float*)d_in[11];
  const float* cls    = (const float*)d_in[12];
  const float* rel_t  = (const float*)d_in[13];
  const int*   rel_i  = (const int*)d_in[14];

  float* out_proj = (float*)d_out;                          // [B][T][C]
  float* aw       = out_proj + (size_t)MM * CC;             // [B][H][T][T]

  char* ws = (char*)d_ws;
  size_t off = 0;
  auto alloc = [&](size_t bytes) {
    void* p = ws + off;
    off = (off + bytes + 255) & ~(size_t)255;
    return p;
  };
  _Float16* xu    = (_Float16*)alloc((size_t)MM * CC * 2);
  _Float16* Wt    = (_Float16*)alloc((size_t)4 * CC * CC * 2);
  _Float16* qws   = (_Float16*)alloc((size_t)BB * HH * NK * HD * 2);
  _Float16* kws   = (_Float16*)alloc((size_t)BB * HH * NK * HD * 2);
  _Float16* vws   = (_Float16*)alloc((size_t)BB * HH * NK * HD * 2);
  _Float16* xattn = (_Float16*)alloc((size_t)MM * CC * 2);
  (void)ws_size; (void)in_sizes; (void)n_in; (void)out_size;

  // prep
  {
    long n = (long)MM * CC;
    prep_xu_kernel<<<(unsigned)((n + 255) / 256), 256, 0, stream>>>(
        x, conv_w, conv_b, pos, cls, xu);
  }
  {
    long n = 4L * CC * CC;
    prep_w_kernel<<<(unsigned)((n + 255) / 256), 256, 0, stream>>>(Wq, Wk, Wv, Wo, Wt);
  }
  {
    long n4 = (3L * BB * HH * NK * HD * 2) / 16;  // q,k,v contiguous
    zero_kernel<<<(unsigned)((n4 + 255) / 256), 256, 0, stream>>>((uint4*)qws, n4);
  }

  // Q/K/V projections (WMMA)
  dim3 ggrid(MM / 32, CC / 64);
  gemm_kernel<<<ggrid, 64, 0, stream>>>(xu, Wt + 0L * CC * CC, bq, qws, 1, QK_SCALE);
  gemm_kernel<<<ggrid, 64, 0, stream>>>(xu, Wt + 1L * CC * CC, bk, kws, 1, 1.0f);
  gemm_kernel<<<ggrid, 64, 0, stream>>>(xu, Wt + 2L * CC * CC, bv, vws, 1, 1.0f);

  // attention (WMMA + LDS softmax)
  size_t smem = (size_t)64 * NK * 2 + 4UL * 16 * NK * 4;  // 233,472 B
  attn_kernel<<<dim3(10, HH, BB), 128, smem, stream>>>(qws, kws, vws, rel_t, rel_i,
                                                       aw, xattn);

  // output projection (WMMA) -> d_out part 1
  gemm_kernel<<<ggrid, 64, 0, stream>>>(xattn, Wt + 3L * CC * CC, bo, out_proj, 0, 1.0f);
}
